// NoStructure_baseline_90512140796595
// MI455X (gfx1250) — matrix operand order repaired and checked by reading, not compiled, not run on hardware
//
#include <hip/hip_runtime.h>

typedef _Float16 v16h __attribute__((ext_vector_type(16)));
typedef _Float16 v8h  __attribute__((ext_vector_type(8)));
typedef float    v8f  __attribute__((ext_vector_type(8)));
typedef int      q4i  __attribute__((vector_size(16)));   // matches builtin's V4i param

#define AS1 __attribute__((address_space(1)))
#define AS3 __attribute__((address_space(3)))

#if defined(__has_builtin)
#if __has_builtin(__builtin_amdgcn_global_load_async_to_lds_b128)
#define HAVE_ASYNC_LDS 1
#endif
#endif
#ifndef HAVE_ASYNC_LDS
#define HAVE_ASYNC_LDS 0
#endif

__device__ __forceinline__ void wait_async0() {
#if __has_builtin(__builtin_amdgcn_s_wait_asynccnt)
  __builtin_amdgcn_s_wait_asynccnt(0);
#else
  asm volatile("s_wait_asynccnt 0" ::: "memory");
#endif
}

// ---------------------------------------------------------------------------
// prep: swizzle W_in (256x256, K-major), Wa1/Wc1 (320x16) into CDNA5 WMMA
// B-fragment layout: [ktile][ntile][lane][16 halfs] -> 32B contiguous per lane.
// B-layout (32x16 f16): lanes 0-15 hold K = kt*32 + 0..7, 16..23 (2 K per VGPR),
// lanes 16-31 hold K = kt*32 + 8..15, 24..31; column N = lane % 16.
// ---------------------------------------------------------------------------
__global__ void prep_kernel(const float* __restrict__ W_in,
                            const float* __restrict__ Wa1,
                            const float* __restrict__ Wc1,
                            _Float16* __restrict__ wB,
                            _Float16* __restrict__ wA1,
                            _Float16* __restrict__ wC1) {
  int e = blockIdx.x * 256 + threadIdx.x;          // 0 .. 65535
  {
    int h  = e & 15;
    int l  = (e >> 4) & 31;
    int nt = (e >> 9) & 15;
    int kt = e >> 13;                              // 0..7
    int k  = kt * 32 + ((l >> 4) << 3) + (h & 7) + ((h >> 3) << 4);  // (l<16 ? 0 : 8) + (h<8 ? h : 8+h)
    int n  = nt * 16 + (l & 15);
    wB[e] = (_Float16)W_in[k * 256 + n];
  }
  if (e < 5120) {                                  // 10 ktiles * 512
    int h  = e & 15;
    int l  = (e >> 4) & 31;
    int kt = e >> 9;                               // 0..9
    int k  = kt * 32 + ((l >> 4) << 3) + (h & 7) + ((h >> 3) << 4);
    int n  = l & 15;
    wA1[e] = (_Float16)Wa1[k * 16 + n];
    wC1[e] = (_Float16)Wc1[k * 16 + n];
  }
}

__device__ __forceinline__ v16h cvt_a_frag(float4 f0, float4 f1, float4 f2, float4 f3) {
  v16h a;
  a[0]=(_Float16)f0.x; a[1]=(_Float16)f0.y; a[2]=(_Float16)f0.z; a[3]=(_Float16)f0.w;
  a[4]=(_Float16)f1.x; a[5]=(_Float16)f1.y; a[6]=(_Float16)f1.z; a[7]=(_Float16)f1.w;
  a[8]=(_Float16)f2.x; a[9]=(_Float16)f2.y; a[10]=(_Float16)f2.z; a[11]=(_Float16)f2.w;
  a[12]=(_Float16)f3.x; a[13]=(_Float16)f3.y; a[14]=(_Float16)f3.z; a[15]=(_Float16)f3.w;
  return a;
}

// ---------------------------------------------------------------------------
// Fused: node_states GEMM (f16 WMMA, f32 accum, async-LDS staged B, double
// buffered) + bias + ReLU, then actor/critic heads from LDS-staged f16 ag.
// 128 threads = 4 waves, 64 rows per workgroup; each wave: 16 rows x 256 cols.
// ---------------------------------------------------------------------------
__global__ __launch_bounds__(128)
void fused_kernel(const float* __restrict__ inputs,
                  const float* __restrict__ goal,
                  const float* __restrict__ b_in,
                  const _Float16* __restrict__ wB,
                  const _Float16* __restrict__ wA1,
                  const _Float16* __restrict__ wC1,
                  const float* __restrict__ ba1, const float* __restrict__ Wa2,
                  const float* __restrict__ ba2,
                  const float* __restrict__ bc1, const float* __restrict__ Wc2,
                  const float* __restrict__ bc2,
                  float* __restrict__ node_out,
                  float* __restrict__ logits,
                  float* __restrict__ crit) {
  __shared__ _Float16 ag[64 * 320];        // 40 KB activations (f16)
  __shared__ _Float16 bstage[2][8192];     // 2 x 16 KB double-buffered B ktile

  const int tid  = threadIdx.x;
  const int lane = tid & 31;
  const int wave = tid >> 5;               // 0..3
  const int half = lane >> 4;              // 0/1
  const int l16  = lane & 15;
  const size_t rowBase = (size_t)blockIdx.x * 64 + wave * 16;

  // stage goal -> ag[:,256:320] as f16 (64 rows x 64 cols / 128 threads)
  #pragma unroll
  for (int it = 0; it < 32; ++it) {
    int flat = tid + 128 * it;             // 0..4095
    int r = flat >> 6, c = flat & 63;
    ag[r * 320 + 256 + c] =
        (_Float16)goal[((size_t)blockIdx.x * 64 + r) * 64 + c];
  }

  // ---- B ktile staging helpers (16 KB = 1024 x 16B chunks / 128 threads) ----
#if !HAVE_ASYNC_LDS
  int4 st[8];
#endif
  auto stage_issue = [&](int kt, int buf) {
    const char* src = (const char*)(wB + (size_t)kt * 8192);
#if HAVE_ASYNC_LDS
    char* dst = (char*)&bstage[buf][0];
    #pragma unroll
    for (int i = 0; i < 8; ++i) {
      int off = (i * 128 + tid) * 16;
      __builtin_amdgcn_global_load_async_to_lds_b128(
          (AS1 q4i*)(src + off), (AS3 q4i*)(dst + off), 0, 0);
    }
#else
    const int4* s = (const int4*)src;
    #pragma unroll
    for (int i = 0; i < 8; ++i) st[i] = s[i * 128 + tid];
#endif
  };
  auto stage_commit = [&](int buf) {
#if HAVE_ASYNC_LDS
    (void)buf;
    wait_async0();
#else
    int4* d = (int4*)&bstage[buf][0];
    #pragma unroll
    for (int i = 0; i < 8; ++i) d[i * 128 + tid] = st[i];
#endif
  };

  // ---- main GEMM: 16 rows x 256 cols per wave, K=256 ----
  v8f acc[16];
  #pragma unroll
  for (int nt = 0; nt < 16; ++nt)
    #pragma unroll
    for (int i = 0; i < 8; ++i) acc[nt][i] = 0.0f;

  const float* arow = inputs + (rowBase + l16) * 256;

  // prologue: stage B ktile 0, prefetch A ktile 0
  stage_issue(0, 0);
  float4 f0 = *(const float4*)(arow + half * 8);
  float4 f1 = *(const float4*)(arow + half * 8 + 4);
  float4 f2 = *(const float4*)(arow + half * 8 + 16);
  float4 f3 = *(const float4*)(arow + half * 8 + 20);
  stage_commit(0);
  __syncthreads();

  for (int kt = 0; kt < 8; ++kt) {
    const int cur = kt & 1;
    if (kt < 7) stage_issue(kt + 1, 1 - cur);      // overlap next B stage
    v16h a = cvt_a_frag(f0, f1, f2, f3);
    if (kt < 7) {                                  // prefetch next A fragment
      int k0 = (kt + 1) * 32 + half * 8;
      f0 = *(const float4*)(arow + k0);
      f1 = *(const float4*)(arow + k0 + 4);
      f2 = *(const float4*)(arow + k0 + 16);
      f3 = *(const float4*)(arow + k0 + 20);
    }
    const _Float16* bbase = &bstage[cur][0];
    v16h b = *(const v16h*)(bbase + (size_t)lane * 16);
    #pragma unroll
    for (int nt = 0; nt < 16; ++nt) {
      v16h bn = b;
      if (nt < 15)                                 // LDS read for nt+1 overlaps WMMA nt
        bn = *(const v16h*)(bbase + ((size_t)(nt + 1) * 32 + lane) * 16);
      acc[nt] = __builtin_amdgcn_wmma_f32_16x16x32_f16(
          false, a, false, b, (short)0, acc[nt], false, false);
      b = bn;
    }
    if (kt < 7) {
      stage_commit(1 - cur);
      __syncthreads();
    }
  }

  // epilogue: bias + ReLU; f32 -> HBM, f16 -> LDS
  #pragma unroll
  for (int nt = 0; nt < 16; ++nt) {
    int col = nt * 16 + l16;
    float bi = b_in[col];
    #pragma unroll
    for (int i = 0; i < 8; ++i) {
      int r = half * 8 + i;                        // C layout: lane<16 -> M=i, else M=8+i
      float v = acc[nt][i] + bi;
      v = v > 0.0f ? v : 0.0f;
      node_out[(rowBase + r) * 256 + col] = v;
      ag[(wave * 16 + r) * 320 + col] = (_Float16)v;
    }
  }
  __syncthreads();

  // ---- heads: K=320, N=16, per-wave 16 rows from LDS ----
  auto head = [&](const _Float16* __restrict__ wH1, const float* __restrict__ bH1,
                  const float* __restrict__ wH2, const float* __restrict__ bH2,
                  float* __restrict__ outv) {
    v8f c;
    #pragma unroll
    for (int i = 0; i < 8; ++i) c[i] = 0.0f;
    #pragma unroll
    for (int kt = 0; kt < 10; ++kt) {
      int k0 = kt * 32 + half * 8;
      const _Float16* ap = &ag[(wave * 16 + l16) * 320 + k0];
      v8h lo = *(const v8h*)ap;
      v8h hi = *(const v8h*)(ap + 16);
      v16h a;
      #pragma unroll
      for (int i = 0; i < 8; ++i) { a[i] = lo[i]; a[8 + i] = hi[i]; }
      v16h b = *(const v16h*)(wH1 + ((size_t)kt * 32 + lane) * 16);
      c = __builtin_amdgcn_wmma_f32_16x16x32_f16(
          false, a, false, b, (short)0, c, false, false);
    }
    float b1 = bH1[l16];
    float w2 = wH2[l16];
    float b2 = bH2[0];
    float sums[8];
    #pragma unroll
    for (int i = 0; i < 8; ++i) {
      float v = c[i] + b1;
      v = v > 0.0f ? v : 0.0f;
      v *= w2;                                     // second linear: dot over 16 cols
      #pragma unroll
      for (int m = 1; m < 16; m <<= 1) v += __shfl_xor(v, m, 32);
      sums[i] = v;
    }
    if (l16 == 0) {
      #pragma unroll
      for (int i = 0; i < 8; ++i)
        outv[rowBase + half * 8 + i] = sums[i] + b2;
    }
  };
  head(wA1, ba1, Wa2, ba2, logits);
  head(wC1, bc1, Wc2, bc2, crit);
}

// ---------------------------------------------------------------------------
// threefry2x32 with key (0,42) == jax.random.key(42), matching jax's
// categorical/gumbel bit stream over shape (1024, 768).
// ---------------------------------------------------------------------------
struct U2 { unsigned x, y; };
__device__ __forceinline__ unsigned rotl32(unsigned v, int r) {
  return (v << r) | (v >> (32 - r));
}
__device__ __forceinline__ U2 threefry42(unsigned x0, unsigned x1) {
  const unsigned k0 = 0u, k1 = 42u, k2 = k0 ^ k1 ^ 0x1BD11BDAu;
  const unsigned ks[3] = {k0, k1, k2};
  const int rot[2][4] = {{13, 15, 26, 6}, {17, 29, 16, 24}};
  x0 += k0; x1 += k1;
  #pragma unroll
  for (int i = 0; i < 5; ++i) {
    const int* r = rot[i & 1];
    #pragma unroll
    for (int j = 0; j < 4; ++j) {
      x0 += x1; x1 = rotl32(x1, r[j]); x1 ^= x0;
    }
    x0 += ks[(i + 1) % 3];
    x1 += ks[(i + 2) % 3] + (unsigned)(i + 1);
  }
  return {x0, x1};
}
__device__ __forceinline__ float gumbel_at(unsigned flat) {   // flat = b*768 + j
  const unsigned H = 393216u;                                 // 1024*768/2
  unsigned c0 = flat < H ? flat : flat - H;
  U2 r = threefry42(c0, c0 + H);
  unsigned bits = flat < H ? r.x : r.y;
  float u = __uint_as_float(0x3f800000u | (bits >> 9)) - 1.0f;
  const float tiny = 1.1754944e-38f;
  u = u * (1.0f - tiny) + tiny;
  u = fmaxf(u, tiny);
  return -logf(-logf(u));
}

// one block per graph: v, entropy, categorical sample, log_prob
__global__ void graph_kernel(const float* __restrict__ logits,
                             const float* __restrict__ crit,
                             const int* __restrict__ nn,
                             float* __restrict__ actions, float* __restrict__ logp,
                             float* __restrict__ ent, float* __restrict__ vout) {
  __shared__ float sf[256];
  __shared__ float sf2[256];
  __shared__ int   si[256];
  const int b = blockIdx.x, t = threadIdx.x;

  // segment offset = prefix sum of num_nodes
  int part = 0;
  for (int j = t; j < b; j += 256) part += nn[j];
  si[t] = part; __syncthreads();
  for (int s = 128; s; s >>= 1) { if (t < s) si[t] += si[t + s]; __syncthreads(); }
  const int offset = si[0];
  const int n = nn[b];
  __syncthreads();

  // critic: 0.5*max + 0.5*mean
  const float* cx = crit + offset;
  float mx = -3.4e38f, sm = 0.0f;
  for (int j = t; j < n; j += 256) { float v = cx[j]; mx = fmaxf(mx, v); sm += v; }
  sf[t] = mx; sf2[t] = sm; __syncthreads();
  for (int s = 128; s; s >>= 1) {
    if (t < s) { sf[t] = fmaxf(sf[t], sf[t + s]); sf2[t] += sf2[t + s]; }
    __syncthreads();
  }
  if (t == 0) vout[b] = 0.5f * sf[0] + 0.5f * (sf2[0] / (float)n);
  __syncthreads();

  // actor softmax: max, sumexp S, T = sum u*e^u  -> ent = log S - T/S
  const float* lx = logits + offset;
  float lm = -3.4e38f;
  for (int j = t; j < n; j += 256) lm = fmaxf(lm, lx[j]);
  sf[t] = lm; __syncthreads();
  for (int s = 128; s; s >>= 1) { if (t < s) sf[t] = fmaxf(sf[t], sf[t + s]); __syncthreads(); }
  const float M = sf[0];
  __syncthreads();

  float S = 0.0f, T = 0.0f;
  for (int j = t; j < n; j += 256) {
    float u = lx[j] - M;
    float e = expf(u);
    S += e; T += u * e;
  }
  sf[t] = S; sf2[t] = T; __syncthreads();
  for (int s = 128; s; s >>= 1) {
    if (t < s) { sf[t] += sf[t + s]; sf2[t] += sf2[t + s]; }
    __syncthreads();
  }
  const float Stot = sf[0], Ttot = sf2[0];
  if (t == 0) ent[b] = logf(Stot) - Ttot / Stot;
  __syncthreads();

  // categorical sample: argmax(logits + gumbel), first index on ties
  float best = -3.4e38f; int bidx = 0x7fffffff;
  for (int j = t; j < n; j += 256) {
    float val = lx[j] + gumbel_at((unsigned)(b * 768 + j));
    if (val > best || (val == best && j < bidx)) { best = val; bidx = j; }
  }
  sf[t] = best; si[t] = bidx; __syncthreads();
  for (int s = 128; s; s >>= 1) {
    if (t < s) {
      float ov = sf[t + s]; int oi = si[t + s];
      if (ov > sf[t] || (ov == sf[t] && oi < si[t])) { sf[t] = ov; si[t] = oi; }
    }
    __syncthreads();
  }
  if (t == 0) {
    int a = si[0];
    actions[b] = (float)a;
    logp[b] = lx[a] - M - logf(Stot);
  }
}

// ---------------------------------------------------------------------------
extern "C" void kernel_launch(void* const* d_in, const int* in_sizes, int n_in,
                              void* d_out, int out_size, void* d_ws, size_t ws_size,
                              hipStream_t stream) {
  const float* inputs = (const float*)d_in[0];
  const float* goal   = (const float*)d_in[1];
  const int*   nnods  = (const int*)d_in[2];
  const float* W_in   = (const float*)d_in[3];
  const float* b_in   = (const float*)d_in[4];
  const float* Wa1    = (const float*)d_in[5];
  const float* ba1    = (const float*)d_in[6];
  const float* Wa2    = (const float*)d_in[7];
  const float* ba2    = (const float*)d_in[8];
  const float* Wc1    = (const float*)d_in[9];
  const float* bc1    = (const float*)d_in[10];
  const float* Wc2    = (const float*)d_in[11];
  const float* bc2    = (const float*)d_in[12];

  const size_t N = (size_t)in_sizes[0] / 256;      // 524288
  const int B = in_sizes[2];                       // 1024

  // workspace layout
  _Float16* wB   = (_Float16*)d_ws;                // 65536 halfs
  _Float16* wA1f = wB + 65536;                     // 5120 halfs
  _Float16* wC1f = wA1f + 5120;                    // 5120 halfs
  float* logits  = (float*)((char*)d_ws + 151552); // N floats
  float* crit    = logits + N;                     // N floats

  // output layout (return order): node_states, actions, log_prob, ent, v
  float* out       = (float*)d_out;
  float* node_out  = out;
  float* actions   = out + N * 256;
  float* logp      = actions + B;
  float* entp      = logp + B;
  float* vp        = entp + B;

  prep_kernel<<<256, 256, 0, stream>>>(W_in, Wa1, Wc1, wB, wA1f, wC1f);
  fused_kernel<<<(int)(N / 64), 128, 0, stream>>>(
      inputs, goal, b_in, wB, wA1f, wC1f,
      ba1, Wa2, ba2, bc1, Wc2, bc2,
      node_out, logits, crit);
  graph_kernel<<<B, 256, 0, stream>>>(logits, crit, nnods, actions, logp, entp, vp);
}
